// MultiheadAttention_51565377356530
// MI455X (gfx1250) — matrix operand order repaired and checked by reading, not compiled, not run on hardware
//
#include <hip/hip_runtime.h>
#include <hip/hip_bf16.h>

// ---------------------------------------------------------------------------
// MHA forward for MI455X (gfx1250, wave32, WMMA, TDM).
// bf16 inputs to v_wmma_f32_16x16x32_bf16, f32 accumulate/softmax/LayerNorm.
// attn_bias (512 MB) streamed once, non-temporal, flash-style online softmax.
// GEMM tiles moved global->LDS by the Tensor Data Mover, double-buffered.
// ---------------------------------------------------------------------------

typedef __bf16 bf;
typedef __attribute__((ext_vector_type(16))) __bf16 v16bf;
typedef __attribute__((ext_vector_type(8)))  __bf16 v8bf;
typedef __attribute__((ext_vector_type(8)))  float  v8f;
typedef __attribute__((ext_vector_type(4)))  float  v4f;
typedef __attribute__((ext_vector_type(4)))  unsigned int u32x4;
typedef __attribute__((ext_vector_type(8)))  int i32x8;
typedef __attribute__((ext_vector_type(4)))  int i32x4;

#define L_SEQ 2048
#define NBATCH 2
#define EMB   1024
#define NH    16
#define HDIM  64
#define BHN   32      // NBATCH*NH
#define MTOT  4096    // L_SEQ*NBATCH

#if defined(__has_builtin)
#if __has_builtin(__builtin_amdgcn_tensor_load_to_lds) && \
    __has_builtin(__builtin_amdgcn_s_wait_tensorcnt)
#define USE_TDM 1
#endif
#endif
#ifndef USE_TDM
#define USE_TDM 0
#endif

__device__ __forceinline__ bf f2bf(float f) {
  unsigned u = __builtin_bit_cast(unsigned, f);
  u += 0x7fffu + ((u >> 16) & 1u);           // round-to-nearest-even
  unsigned short h = (unsigned short)(u >> 16);
  return __builtin_bit_cast(bf, h);
}

__device__ __forceinline__ v16bf cat16(v8bf lo, v8bf hi) {
  return __builtin_shufflevector(lo, hi, 0,1,2,3,4,5,6,7,8,9,10,11,12,13,14,15);
}

__device__ __forceinline__ v8f wmma_bf16(v16bf a, v16bf b, v8f c) {
  return __builtin_amdgcn_wmma_f32_16x16x32_bf16(false, a, false, b, (short)0, c,
                                                 false, false);
}

__device__ __forceinline__ v8f vzero8() {
  v8f z = {0.f,0.f,0.f,0.f,0.f,0.f,0.f,0.f};
  return z;
}

#if USE_TDM
// Issue one TDM 2-D tile load: 64x64 bf16 tile, row stride `stride_e` elements,
// packed row-major into LDS at `lds_dst`. Descriptor per ISA 08_async_tensor §8.
__device__ __forceinline__ void tdm_load_tile64(const bf* gsrc, bf* lds_dst,
                                                int tensor_rows, int stride_e) {
  unsigned lds_addr = (unsigned)(size_t)lds_dst;   // low 32b of generic = LDS off
  unsigned long long ga = (unsigned long long)(size_t)gsrc;
  u32x4 g0;
  g0.x = 1u;                                        // count=1, is_restore=0
  g0.y = lds_addr;                                  // bits 63:32
  g0.z = (unsigned)ga;                              // global_addr lo
  g0.w = (unsigned)((ga >> 32) & 0x01FFFFFFull) | (2u << 30);  // ga hi | type=2
  i32x8 g1;
  g1[0] = 0x00010000;                               // wg_mask=0, data_size=1 (2B)
  g1[1] = (int)(((unsigned)stride_e & 0xFFFFu) << 16);         // tensor_dim0 lo16
  g1[2] = (int)((((unsigned)stride_e >> 16) & 0xFFFFu) |
                (((unsigned)tensor_rows & 0xFFFFu) << 16));    // dim0 hi | dim1 lo
  g1[3] = (int)((((unsigned)tensor_rows >> 16) & 0xFFFFu) |
                (64u << 16));                       // dim1 hi | tile_dim0=64
  g1[4] = 64;                                       // tile_dim1=64, tile_dim2=0
  g1[5] = stride_e;                                 // tensor_dim0_stride lo32
  g1[6] = 0;
  g1[7] = 0;
  i32x4 z4 = {0, 0, 0, 0};
#if __clang_major__ >= 23
  i32x8 z8 = {0, 0, 0, 0, 0, 0, 0, 0};
  __builtin_amdgcn_tensor_load_to_lds(g0, g1, z4, z4, z8, 0);
#else
  __builtin_amdgcn_tensor_load_to_lds(g0, g1, z4, z4, 0);
#endif
}
#endif

// ---------------------------------------------------------------------------
// query f32 -> bf16 (same flat layout [L,B,E])
// ---------------------------------------------------------------------------
__global__ __launch_bounds__(256)
void k_cvt_query(const float* __restrict__ q, bf* __restrict__ qbf) {
  int i = (blockIdx.x * 256 + threadIdx.x) * 4;
  v4f v = *(const v4f*)(q + i);
  qbf[i + 0] = f2bf(v.x);
  qbf[i + 1] = f2bf(v.y);
  qbf[i + 2] = f2bf(v.z);
  qbf[i + 3] = f2bf(v.w);
}

// ---------------------------------------------------------------------------
// Weight f32 [n][k] -> bf16 transposed [k][n]  (B-major for WMMA)
// ---------------------------------------------------------------------------
__global__ __launch_bounds__(256)
void k_cvt_wT(const float* __restrict__ Wq, const float* __restrict__ Wk,
              const float* __restrict__ Wv, const float* __restrict__ Wo,
              bf* __restrict__ wbf) {
  __shared__ float t[16][17];
  const float* W = (blockIdx.z == 0) ? Wq :
                   (blockIdx.z == 1) ? Wk :
                   (blockIdx.z == 2) ? Wv : Wo;
  int tx = threadIdx.x, ty = threadIdx.y;
  int n0 = blockIdx.y * 16, k0 = blockIdx.x * 16;
  t[ty][tx] = W[(size_t)(n0 + ty) * EMB + (k0 + tx)];
  __syncthreads();
  bf* out = wbf + (size_t)blockIdx.z * EMB * EMB;
  out[(size_t)(k0 + ty) * EMB + (n0 + tx)] = f2bf(t[tx][ty]);
}

// ---------------------------------------------------------------------------
// Tiled WMMA GEMM: C[M][1024] = A[M][1024] * Bt[1024][1024]
// 64x64 block tile, 128 threads (4 waves), each wave a 16x64 strip.
// Double-buffered LDS tiles filled by the Tensor Data Mover (wave 0 issues,
// s_wait_tensorcnt + barrier publishes); manual-copy fallback if no TDM builtin.
// Epilogue modes:
//   0: Q  -> (acc+bq)*HD^-0.5, bf16 to Qh [bh][l][d]
//   1: Kt -> bf16 to Kt [bh][d][l]  (LDS-staged transpose store)
//   2: V  -> (acc+bv),  bf16 to Vh [bh][l][d]
//   3: out-> (acc+bo),  f32 to d_out [L,B,E]
// ---------------------------------------------------------------------------
__global__ __launch_bounds__(128)
void k_gemm(const bf* __restrict__ A, const bf* __restrict__ Bt,
            const float* __restrict__ bias, void* __restrict__ outp,
            int mode, int mrows) {
  __shared__ __align__(16) bf As[2][64][64];
  __shared__ __align__(16) bf Bs[2][64][64];
  __shared__ __align__(16) bf Stage[4][64][16];

  const int tid  = threadIdx.x;
  const int w    = tid >> 5;
  const int lane = tid & 31;
  const int ml   = lane & 15;
  const int hf   = lane >> 4;
  const int mblk = blockIdx.y, nblk = blockIdx.x;

  v8f acc[4];
  #pragma unroll
  for (int nf = 0; nf < 4; ++nf) acc[nf] = vzero8();

  const bf* Arow = A + (size_t)(mblk * 64) * EMB;       // tile row base
  const bf* Bcol = Bt + nblk * 64;                      // tile col base

#if USE_TDM
  // prologue: fetch tile 0 into buffer 0
  if (tid < 32) {
    tdm_load_tile64(Arow, &As[0][0][0], mrows, EMB);
    tdm_load_tile64(Bcol, &Bs[0][0][0], EMB, EMB);
  }
  __builtin_amdgcn_s_wait_tensorcnt(0);
  __syncthreads();
#endif

  int buf = 0;
  for (int kb = 0; kb < 16; ++kb) {
#if USE_TDM
    // async prefetch of next K-tile into the other buffer while we compute
    if (kb + 1 < 16 && tid < 32) {
      tdm_load_tile64(Arow + (kb + 1) * 64, &As[buf ^ 1][0][0], mrows, EMB);
      tdm_load_tile64(Bcol + (size_t)(kb + 1) * 64 * EMB, &Bs[buf ^ 1][0][0],
                      EMB, EMB);
    }
#else
    #pragma unroll
    for (int i = 0; i < 4; ++i) {
      int lin = tid + i * 128;
      int r = lin >> 3, c8 = (lin & 7) * 8;
      *(v8bf*)&As[buf][r][c8] = *(const v8bf*)(Arow + (size_t)r * EMB + kb * 64 + c8);
      *(v8bf*)&Bs[buf][r][c8] = *(const v8bf*)(Bcol + (size_t)(kb * 64 + r) * EMB + c8);
    }
    __syncthreads();
#endif

    // Preload all fragments, then burst the WMMAs (amortizes s_wait_dscnt).
    v16bf af[2], bfr[2][4];
    #pragma unroll
    for (int s = 0; s < 2; ++s) {
      af[s] = cat16(*(const v8bf*)&As[buf][w * 16 + ml][s * 32 + hf * 8],
                    *(const v8bf*)&As[buf][w * 16 + ml][s * 32 + 16 + hf * 8]);
      #pragma unroll
      for (int nf = 0; nf < 4; ++nf)
        #pragma unroll
        for (int e = 0; e < 16; ++e)
          bfr[s][nf][e] = Bs[buf][s * 32 + (e >> 3) * 16 + hf * 8 + (e & 7)][nf * 16 + ml];
    }
    #pragma unroll
    for (int s = 0; s < 2; ++s)
      #pragma unroll
      for (int nf = 0; nf < 4; ++nf)
        acc[nf] = wmma_bf16(af[s], bfr[s][nf], acc[nf]);

#if USE_TDM
    __builtin_amdgcn_s_wait_tensorcnt(0);   // next tile landed
    __syncthreads();                        // publish to all waves / recycle buf
    buf ^= 1;
#else
    __syncthreads();
#endif
  }

  if (mode == 3) {                                   // output projection, f32
    float* out = (float*)outp;
    #pragma unroll
    for (int nf = 0; nf < 4; ++nf)
      #pragma unroll
      for (int i = 0; i < 8; ++i) {
        int tok = mblk * 64 + w * 16 + i + hf * 8;   // row = l*B + b
        int gn  = nblk * 64 + nf * 16 + ml;
        out[(size_t)tok * EMB + gn] = acc[nf][i] + bias[gn];
      }
  } else if (mode == 0 || mode == 2) {               // Q or V, head-major bf16
    bf* out = (bf*)outp;
    const float scale = (mode == 0) ? 0.125f : 1.0f; // HD^-0.5 = 1/8
    #pragma unroll
    for (int nf = 0; nf < 4; ++nf)
      #pragma unroll
      for (int i = 0; i < 8; ++i) {
        int tok = mblk * 64 + w * 16 + i + hf * 8;
        int b = tok & 1, l = tok >> 1;
        int d = nf * 16 + ml;                        // head h == nblk
        float bval = bias ? bias[nblk * 64 + d] : 0.f;
        out[((size_t)(b * NH + nblk) * L_SEQ + l) * HDIM + d] =
            f2bf((acc[nf][i] + bval) * scale);
      }
  } else {                                           // mode 1: K transposed
    #pragma unroll
    for (int nf = 0; nf < 4; ++nf)
      #pragma unroll
      for (int i = 0; i < 8; ++i) {
        int tl = i + hf * 8;                         // 0..15 within strip
        int d  = nf * 16 + ml;
        Stage[w][d][(tl & 1) * 8 + (tl >> 1)] = f2bf(acc[nf][i]);
      }
    bf* out = (bf*)outp;
    int base = mblk * 64 + w * 16;                   // even
    int l0 = base >> 1;
    #pragma unroll
    for (int j = 0; j < 2; ++j) {
      int idx = j * 32 + lane;                       // 0..63
      int d = idx >> 1, b = idx & 1;
      bf* dst = out + ((size_t)(b * NH + nblk) * HDIM + d) * L_SEQ + l0;
      *(v8bf*)dst = *(const v8bf*)&Stage[w][d][b * 8];
    }
  }
}

// ---------------------------------------------------------------------------
// Flash attention: grid (L/64 q-tiles, 32 heads), 128 threads (4 waves),
// each wave owns a 16-row strip. Online softmax; bias streamed NT once.
// ---------------------------------------------------------------------------
__global__ __launch_bounds__(128)
void k_attn(const bf* __restrict__ Qh, const bf* __restrict__ Kt,
            const bf* __restrict__ Vh, const float* __restrict__ bias,
            float* __restrict__ attnf) {
  __shared__ __align__(16) bf Plds[4][16][64];

  const int tid  = threadIdx.x;
  const int w    = tid >> 5;
  const int lane = tid & 31;
  const int ml   = lane & 15;
  const int hf   = lane >> 4;
  const int bh   = blockIdx.y;
  const int q0   = blockIdx.x * 64 + w * 16;         // l base of this strip

  // Q fragments (16x64, K split into two 32-steps), already scaled by 1/8
  const bf* Qp = Qh + ((size_t)bh * L_SEQ + q0 + ml) * HDIM;
  v16bf a0 = cat16(*(const v8bf*)(Qp + hf * 8),
                   *(const v8bf*)(Qp + 16 + hf * 8));
  v16bf a1 = cat16(*(const v8bf*)(Qp + 32 + hf * 8),
                   *(const v8bf*)(Qp + 48 + hf * 8));

  v8f o[4];
  #pragma unroll
  for (int nf = 0; nf < 4; ++nf) o[nf] = vzero8();
  float mrow[8], lrow[8];
  #pragma unroll
  for (int i = 0; i < 8; ++i) { mrow[i] = -INFINITY; lrow[i] = 0.f; }

  const float* bptr = bias + ((size_t)bh * L_SEQ + q0) * L_SEQ;

  for (int kt = 0; kt < 32; ++kt) {
    const int kbase = kt * 64;
    if (kt + 1 < 32) __builtin_prefetch(bptr + kbase + 64, 0, 0);

    // ---- scores S = Q * K^T  (Kt layout [bh][d][l]: lane=n=l, elements over k=d)
    v8f s[4];
    #pragma unroll
    for (int nf = 0; nf < 4; ++nf) {
      const bf* Kp = Kt + (size_t)bh * HDIM * L_SEQ + kbase + nf * 16 + ml;
      v16bf b0, b1;
      #pragma unroll
      for (int e = 0; e < 16; ++e) {
        int d = (e >> 3) * 16 + hf * 8 + (e & 7);
        b0[e] = Kp[(size_t)d * L_SEQ];
        b1[e] = Kp[(size_t)(32 + d) * L_SEQ];
      }
      v8f z = vzero8();
      z = wmma_bf16(a0, b0, z);
      s[nf] = wmma_bf16(a1, b1, z);
    }

    // ---- + bias (non-temporal single-use stream, 512 MB total)
    #pragma unroll
    for (int nf = 0; nf < 4; ++nf)
      #pragma unroll
      for (int i = 0; i < 8; ++i) {
        int r = i + hf * 8;
        s[nf][i] += __builtin_nontemporal_load(
            bptr + (size_t)r * L_SEQ + kbase + nf * 16 + ml);
      }

    // ---- online softmax (row lives across a 16-lane half)
    float cm[8];
    #pragma unroll
    for (int i = 0; i < 8; ++i)
      cm[i] = fmaxf(fmaxf(s[0][i], s[1][i]), fmaxf(s[2][i], s[3][i]));
    #pragma unroll
    for (int msk = 8; msk >= 1; msk >>= 1)
      #pragma unroll
      for (int i = 0; i < 8; ++i)
        cm[i] = fmaxf(cm[i], __shfl_xor(cm[i], msk, 32));

    float al[8];
    #pragma unroll
    for (int i = 0; i < 8; ++i) {
      float nm = fmaxf(mrow[i], cm[i]);
      al[i] = __expf(mrow[i] - nm);
      mrow[i] = nm;
    }

    float ps[8];
    #pragma unroll
    for (int i = 0; i < 8; ++i) ps[i] = 0.f;
    #pragma unroll
    for (int nf = 0; nf < 4; ++nf)
      #pragma unroll
      for (int i = 0; i < 8; ++i) {
        float p = __expf(s[nf][i] - mrow[i]);
        ps[i] += p;
        Plds[w][i + hf * 8][nf * 16 + ml] = f2bf(p);
      }
    #pragma unroll
    for (int msk = 8; msk >= 1; msk >>= 1)
      #pragma unroll
      for (int i = 0; i < 8; ++i)
        ps[i] += __shfl_xor(ps[i], msk, 32);
    #pragma unroll
    for (int i = 0; i < 8; ++i) lrow[i] = lrow[i] * al[i] + ps[i];
    #pragma unroll
    for (int nf = 0; nf < 4; ++nf)
      #pragma unroll
      for (int i = 0; i < 8; ++i) o[nf][i] *= al[i];

    // ---- O += P * V   (P restaged via LDS into A-fragment layout)
    v16bf p0 = cat16(*(const v8bf*)&Plds[w][ml][hf * 8],
                     *(const v8bf*)&Plds[w][ml][16 + hf * 8]);
    v16bf p1 = cat16(*(const v8bf*)&Plds[w][ml][32 + hf * 8],
                     *(const v8bf*)&Plds[w][ml][48 + hf * 8]);
    #pragma unroll
    for (int nf = 0; nf < 4; ++nf) {
      const bf* Vp = Vh + ((size_t)bh * L_SEQ + kbase) * HDIM + nf * 16 + ml;
      v16bf b0, b1;
      #pragma unroll
      for (int e = 0; e < 16; ++e) {
        int kk = (e >> 3) * 16 + hf * 8 + (e & 7);
        b0[e] = Vp[(size_t)kk * HDIM];
        b1[e] = Vp[(size_t)(32 + kk) * HDIM];
      }
      o[nf] = wmma_bf16(p0, b0, o[nf]);
      o[nf] = wmma_bf16(p1, b1, o[nf]);
    }
  }

  // ---- normalize and write attn (pre-LN) f32 [L,B,E]
  const int b = bh >> 4, h = bh & 15;
  #pragma unroll
  for (int i = 0; i < 8; ++i) {
    float inv = 1.f / lrow[i];
    #pragma unroll
    for (int nf = 0; nf < 4; ++nf) {
      int l = q0 + i + hf * 8;
      attnf[(size_t)(l * NBATCH + b) * EMB + h * HDIM + nf * 16 + ml] =
          o[nf][i] * inv;
    }
  }
}

// ---------------------------------------------------------------------------
// LayerNorm over E=1024, one row per block; writes bf16 for the final GEMM.
// ---------------------------------------------------------------------------
__global__ __launch_bounds__(256)
void k_ln(const float* __restrict__ x, const float* __restrict__ g,
          const float* __restrict__ bta, bf* __restrict__ y) {
  __shared__ float rs[8], rq[8];
  const int r = blockIdx.x;
  const int lane = threadIdx.x & 31, wid = threadIdx.x >> 5;
  const float* xr = x + (size_t)r * EMB;
  float v[4], sum = 0.f, sq = 0.f;
  #pragma unroll
  for (int j = 0; j < 4; ++j) {
    v[j] = xr[threadIdx.x + j * 256];
    sum += v[j];
    sq  += v[j] * v[j];
  }
  #pragma unroll
  for (int m = 16; m >= 1; m >>= 1) {
    sum += __shfl_xor(sum, m, 32);
    sq  += __shfl_xor(sq,  m, 32);
  }
  if (lane == 0) { rs[wid] = sum; rq[wid] = sq; }
  __syncthreads();
  float ts = 0.f, tq = 0.f;
  #pragma unroll
  for (int i = 0; i < 8; ++i) { ts += rs[i]; tq += rq[i]; }
  const float mean = ts * (1.f / EMB);
  const float var  = tq * (1.f / EMB) - mean * mean;
  const float rstd = rsqrtf(var + 1e-5f);
  #pragma unroll
  for (int j = 0; j < 4; ++j) {
    int e = threadIdx.x + j * 256;
    y[(size_t)r * EMB + e] = f2bf((v[j] - mean) * rstd * g[e] + bta[e]);
  }
}

// ---------------------------------------------------------------------------
extern "C" void kernel_launch(void* const* d_in, const int* in_sizes, int n_in,
                              void* d_out, int out_size, void* d_ws, size_t ws_size,
                              hipStream_t stream) {
  const float* query = (const float*)d_in[0];
  const float* attn_bias = (const float*)d_in[1];
  const float* Wq = (const float*)d_in[2];
  const float* bq = (const float*)d_in[3];
  const float* Wk = (const float*)d_in[4];
  const float* Wv = (const float*)d_in[5];
  const float* bv = (const float*)d_in[6];
  const float* Wo = (const float*)d_in[7];
  const float* bo = (const float*)d_in[8];
  const float* ln_g = (const float*)d_in[9];
  const float* ln_b = (const float*)d_in[10];

  char* ws = (char*)d_ws;
  bf*    qbf    = (bf*)(ws + (size_t)0);             //  8 MB: query bf16
  bf*    wbf    = (bf*)(ws + ((size_t)8  << 20));    //  8 MB: 4 weights [k][n]
  bf*    Qh     = (bf*)(ws + ((size_t)16 << 20));    //  8 MB: Q  [bh][l][d]
  bf*    Ktb    = (bf*)(ws + ((size_t)24 << 20));    //  8 MB: K  [bh][d][l]
  bf*    Vhb    = (bf*)(ws + ((size_t)32 << 20));    //  8 MB: V  [bh][l][d]
  float* attnf  = (float*)(ws + ((size_t)40 << 20)); // 16 MB: attn pre-LN f32
  bf*    attnbf = (bf*)(ws + ((size_t)56 << 20));    //  8 MB: attn post-LN bf16

  k_cvt_query<<<4096, 256, 0, stream>>>(query, qbf);
  k_cvt_wT<<<dim3(64, 64, 4), dim3(16, 16), 0, stream>>>(Wq, Wk, Wv, Wo, wbf);

  const size_t WSTRIDE = (size_t)EMB * EMB;          // elements per weight
  dim3 gg(16, 64);                                    // N/64 x M/64
  k_gemm<<<gg, 128, 0, stream>>>(qbf, wbf + 0 * WSTRIDE, bq,      Qh,  0, MTOT);
  k_gemm<<<gg, 128, 0, stream>>>(qbf, wbf + 1 * WSTRIDE, nullptr, Ktb, 1, MTOT);
  k_gemm<<<gg, 128, 0, stream>>>(qbf, wbf + 2 * WSTRIDE, bv,      Vhb, 2, MTOT);

  k_attn<<<dim3(L_SEQ / 64, BHN), 128, 0, stream>>>(Qh, Ktb, Vhb, attn_bias, attnf);

  k_ln<<<MTOT, 256, 0, stream>>>(attnf, ln_g, ln_b, attnbf);

  k_gemm<<<gg, 128, 0, stream>>>(attnbf, wbf + 3 * WSTRIDE, bo, d_out, 3, MTOT);
}
